// DCEModulatedResBlock_89429809037444
// MI455X (gfx1250) — compile-verified, matrix operand order checked by reading, not run on hardware
//
#include <hip/hip_runtime.h>

typedef unsigned short u16;
typedef __attribute__((ext_vector_type(16))) __bf16 v16bf;
typedef __attribute__((ext_vector_type(8)))  float  v8f;

struct FragBits { uint4 lo, hi; };

__device__ __forceinline__ v16bf frag_cast(uint4 lo, uint4 hi) {
  FragBits f{lo, hi};
  return __builtin_bit_cast(v16bf, f);
}

__device__ __forceinline__ u16 f2bf(float f) {
  unsigned u = __float_as_uint(f);
  u += 0x7fffu + ((u >> 16) & 1u);   // round-to-nearest-even bf16
  return (u16)(u >> 16);
}

__device__ __forceinline__ float sigmoidf_(float z) { return 1.f / (1.f + expf(-z)); }

// ---------------------------------------------------------------------------
// 1. DCE pooling + projection: (B,100,128) -> mean over seq -> Linear -> (B,256)
// ---------------------------------------------------------------------------
__global__ __launch_bounds__(256)
void k_dce(const float* __restrict__ dce, const float* __restrict__ wd,
           const float* __restrict__ bd, float* __restrict__ proj) {
  __shared__ float pooled[128];
  int b = blockIdx.x, tid = threadIdx.x;
  if (tid < 128) {
    float s = 0.f;
    const float* base = dce + (size_t)b * 100 * 128 + tid;
    for (int i = 0; i < 100; ++i) s += base[i * 128];
    pooled[tid] = s * 0.01f;
  }
  __syncthreads();
  float s = bd[tid];
  const float* wrow = wd + tid * 128;
  for (int j = 0; j < 128; ++j) s += wrow[j] * pooled[j];
  proj[b * 256 + tid] = s;
}

// ---------------------------------------------------------------------------
// 2. Depthwise 3x3 conv + global average pool -> spatial_proj (B,256)
// ---------------------------------------------------------------------------
__global__ __launch_bounds__(256)
void k_spatial(const float* __restrict__ x, const float* __restrict__ wch,
               float* __restrict__ spat) {
  __shared__ float red[256];
  int bc = blockIdx.x;          // b*256 + c
  int c = bc & 255;
  const float* plane = x + (size_t)bc * 4096;
  float w[9];
#pragma unroll
  for (int i = 0; i < 9; ++i) w[i] = wch[c * 9 + i];
  float s = 0.f;
  for (int pix = threadIdx.x; pix < 4096; pix += 256) {
    int h = pix >> 6, wc = pix & 63;
    float a = 0.f;
#pragma unroll
    for (int ky = 0; ky < 3; ++ky)
#pragma unroll
      for (int kx = 0; kx < 3; ++kx) {
        int h2 = h + ky - 1, w2 = wc + kx - 1;
        if (((unsigned)h2 < 64u) & ((unsigned)w2 < 64u))
          a += w[ky * 3 + kx] * plane[(h2 << 6) + w2];
      }
    s += a;
  }
  red[threadIdx.x] = s;
  __syncthreads();
  for (int st = 128; st > 0; st >>= 1) {
    if (threadIdx.x < st) red[threadIdx.x] += red[threadIdx.x + st];
    __syncthreads();
  }
  if (threadIdx.x == 0) spat[bc] = red[0] * (1.f / 4096.f);
}

// ---------------------------------------------------------------------------
// 3. Gate MLP: m = proj*spat; h = relu(m@Ws^T+bs); mod = sigmoid(h@We^T+be)
// ---------------------------------------------------------------------------
__global__ __launch_bounds__(256)
void k_gate(const float* __restrict__ proj, const float* __restrict__ spat,
            const float* __restrict__ wsk, const float* __restrict__ bsk,
            const float* __restrict__ wex, const float* __restrict__ bex,
            float* __restrict__ mod) {
  __shared__ float m[256];
  __shared__ float hbuf[128];
  int b = blockIdx.x, tid = threadIdx.x;
  m[tid] = proj[b * 256 + tid] * spat[b * 256 + tid];
  __syncthreads();
  if (tid < 128) {
    float s = bsk[tid];
    const float* wr = wsk + tid * 256;
    for (int j = 0; j < 256; ++j) s += wr[j] * m[j];
    hbuf[tid] = fmaxf(s, 0.f);
  }
  __syncthreads();
  float s = bex[tid];
  const float* wr = wex + tid * 128;
  for (int j = 0; j < 128; ++j) s += wr[j] * hbuf[j];
  mod[b * 256 + tid] = sigmoidf_(s);
}

// ---------------------------------------------------------------------------
// 4. Weight repack fp32 -> bf16 GEMM layout.
//    wb1[co][tap][ci]   (K = tap*256+ci),  wb2/wbs[co][ci]
// ---------------------------------------------------------------------------
__global__ __launch_bounds__(256)
void k_prep(const float* __restrict__ w1, const float* __restrict__ w2,
            const float* __restrict__ wsc, u16* __restrict__ wb1,
            u16* __restrict__ wb2, u16* __restrict__ wbs) {
  int i = blockIdx.x * 256 + threadIdx.x;
  if (i < 256 * 9 * 256) {
    int co = i / 2304, r = i % 2304, t = r >> 8, ci = r & 255;
    wb1[i] = f2bf(w1[(co * 256 + ci) * 9 + t]);     // OIHW -> [co][t][ci]
  }
  if (i < 65536) { wb2[i] = f2bf(w2[i]); wbs[i] = f2bf(wsc[i]); }
}

// ---------------------------------------------------------------------------
// 5. xm = x * mod, NCHW fp32 -> NHWC bf16 via LDS tile transpose
// ---------------------------------------------------------------------------
__global__ __launch_bounds__(256)
void k_xm(const float* __restrict__ x, const float* __restrict__ mod,
          u16* __restrict__ xm) {
  __shared__ float t[32][33];
  int n = blockIdx.z, c0 = blockIdx.y << 5, hw0 = blockIdx.x << 5;
  int tx = threadIdx.x & 31, ty = threadIdx.x >> 5;
#pragma unroll
  for (int j = 0; j < 4; ++j) {
    int cc = (ty << 2) + j;
    float v = x[((size_t)(n * 256 + c0 + cc) << 12) + hw0 + tx] *
              mod[n * 256 + c0 + cc];
    t[cc][tx] = v;                               // [channel][pixel]
  }
  __syncthreads();
#pragma unroll
  for (int j = 0; j < 4; ++j) {
    int pp = (ty << 2) + j;
    xm[((size_t)((n << 12) + hw0 + pp) << 8) + c0 + tx] = f2bf(t[tx][pp]);
  }
}

// ---------------------------------------------------------------------------
// 6. Implicit-GEMM conv via WMMA bf16 (taps=9 -> 3x3 SAME; taps=1 -> 1x1)
//    A: NHWC bf16 [65536][256]   Bw: [256][taps*256] bf16   Y: fp32 [65536][256]
//    Block tile 128x128, BK=32; 8 waves of 64x32 (4x2 WMMA frags).
// ---------------------------------------------------------------------------
__global__ __launch_bounds__(256)
void k_gemm(const u16* __restrict__ A, const u16* __restrict__ Bw,
            float* __restrict__ Y, int taps) {
  __shared__ __align__(16) u16 tA[128 * 40];   // padded stride: bank-conflict-free
  __shared__ __align__(16) u16 tB[128 * 40];
  const int tid  = threadIdx.x;
  const int lane = tid & 31;
  const int wv   = tid >> 5;
  const int wm   = (wv & 1) << 6;     // wave M offset: 0/64
  const int wn   = (wv >> 1) << 5;    // wave N offset: 0/32/64/96
  const int m0   = blockIdx.x << 7;
  const int n0   = blockIdx.y << 7;
  const int KT   = taps << 8;
  const int iters = KT >> 5;

  // global->LDS staging: 2 threads per tile row, 32B each
  const int lrow  = tid >> 1;
  const int lhalf = tid & 1;
  const int p    = m0 + lrow;
  const int nimg = p >> 12;
  const int hh   = (p >> 6) & 63;
  const int ww   = p & 63;

  auto loadTiles = [&](int it, uint4& ra0, uint4& ra1, uint4& rb0, uint4& rb1) {
    int dy = 0, dx = 0;
    int ci0 = (it & 7) << 5;                   // 8 K-iters per tap
    if (taps == 9) { int t = it >> 3; dy = t / 3 - 1; dx = t % 3 - 1; }
    int h2 = hh + dy, w2 = ww + dx;
    if (((unsigned)h2 < 64u) & ((unsigned)w2 < 64u)) {
      const u16* src = A + ((size_t)((nimg << 12) + (h2 << 6) + w2) * 256 +
                            ci0 + (lhalf << 4));
      ra0 = *(const uint4*)src;
      ra1 = *(const uint4*)(src + 8);
    } else {
      ra0 = make_uint4(0, 0, 0, 0);
      ra1 = make_uint4(0, 0, 0, 0);
    }
    const u16* bsrc = Bw + ((size_t)(n0 + lrow) * KT + (it << 5) + (lhalf << 4));
    rb0 = *(const uint4*)bsrc;
    rb1 = *(const uint4*)(bsrc + 8);
  };

  uint4 a0, a1, b0, b1;
  loadTiles(0, a0, a1, b0, b1);

  v8f acc[4][2] = {};
  const int frow = lane & 15;          // M (A) / N (B) row within fragment
  const int kb   = (lane >> 4) << 3;   // K chunk base per ISA 16-bit layout

  for (int it = 0; it < iters; ++it) {
    *(uint4*)&tA[lrow * 40 + (lhalf << 4)]     = a0;
    *(uint4*)&tA[lrow * 40 + (lhalf << 4) + 8] = a1;
    *(uint4*)&tB[lrow * 40 + (lhalf << 4)]     = b0;
    *(uint4*)&tB[lrow * 40 + (lhalf << 4) + 8] = b1;
    __syncthreads();

    uint4 na0 = a0, na1 = a1, nb0 = b0, nb1 = b1;
    if (it + 1 < iters) loadTiles(it + 1, na0, na1, nb0, nb1);
    if (it + 2 < iters)  // gfx1250 global_prefetch_b8 on upcoming weight tile
      __builtin_prefetch(Bw + ((size_t)(n0 + lrow) * KT + ((it + 2) << 5)), 0, 0);

    v16bf af[4], bfr[2];
#pragma unroll
    for (int mf = 0; mf < 4; ++mf) {
      int r = wm + (mf << 4) + frow;
      af[mf] = frag_cast(*(const uint4*)&tA[r * 40 + kb],
                         *(const uint4*)&tA[r * 40 + kb + 16]);
    }
#pragma unroll
    for (int nf = 0; nf < 2; ++nf) {
      int r = wn + (nf << 4) + frow;
      bfr[nf] = frag_cast(*(const uint4*)&tB[r * 40 + kb],
                          *(const uint4*)&tB[r * 40 + kb + 16]);
    }
#pragma unroll
    for (int mf = 0; mf < 4; ++mf)
#pragma unroll
      for (int nf = 0; nf < 2; ++nf)
        acc[mf][nf] = __builtin_amdgcn_wmma_f32_16x16x32_bf16(
            false, af[mf], false, bfr[nf], (short)0, acc[mf][nf], false, false);
    __syncthreads();
    a0 = na0; a1 = na1; b0 = nb0; b1 = nb1;
  }

  // Epilogue: C/D layout VGPR i -> M=i (lanes 0-15), M=i+8 (lanes 16-31)
  const int csel = (lane >> 4) << 3;
#pragma unroll
  for (int mf = 0; mf < 4; ++mf) {
    int rbase = m0 + wm + (mf << 4) + csel;
#pragma unroll
    for (int nf = 0; nf < 2; ++nf) {
      int col = n0 + wn + (nf << 4) + frow;
#pragma unroll
      for (int i = 0; i < 8; ++i)
        Y[(size_t)(rbase + i) * 256 + col] = acc[mf][nf][i];
    }
  }
}

// ---------------------------------------------------------------------------
// 7. Deterministic two-pass BN stats over NHWC fp32 [65536][256]
// ---------------------------------------------------------------------------
__global__ __launch_bounds__(256)
void k_statsp(const float* __restrict__ Y, float* __restrict__ ps,
              float* __restrict__ pq) {
  int tid = threadIdx.x;
  size_t r0 = (size_t)blockIdx.x * 256;
  float s = 0.f, q = 0.f;
  for (int r = 0; r < 256; ++r) {
    float v = Y[(r0 + r) * 256 + tid];
    s += v; q += v * v;
  }
  ps[blockIdx.x * 256 + tid] = s;
  pq[blockIdx.x * 256 + tid] = q;
}

__global__ __launch_bounds__(256)
void k_statsf(const float* __restrict__ ps, const float* __restrict__ pq,
              const float* __restrict__ g, const float* __restrict__ be,
              float* __restrict__ scale, float* __restrict__ shift) {
  __shared__ float rs[256], rq[256];
  int c = blockIdx.x, tid = threadIdx.x;
  rs[tid] = ps[tid * 256 + c];
  rq[tid] = pq[tid * 256 + c];
  __syncthreads();
  for (int st = 128; st > 0; st >>= 1) {
    if (tid < st) { rs[tid] += rs[tid + st]; rq[tid] += rq[tid + st]; }
    __syncthreads();
  }
  if (tid == 0) {
    float mean = rs[0] * (1.f / 65536.f);
    float var  = rq[0] * (1.f / 65536.f) - mean * mean;
    float s = g[c] * rsqrtf(var + 1e-5f);
    scale[c] = s;
    shift[c] = be[c] - mean * s;
  }
}

// ---------------------------------------------------------------------------
// 8. BN + SiLU, fp32 NHWC -> bf16 NHWC (input to conv2)
// ---------------------------------------------------------------------------
__global__ __launch_bounds__(256)
void k_bnact(const float* __restrict__ Y, const float* __restrict__ sc,
             const float* __restrict__ sh, u16* __restrict__ out) {
  int i = blockIdx.x * 256 + threadIdx.x;
  int c = i & 255;
  float z = Y[i] * sc[c] + sh[c];
  out[i] = f2bf(z * sigmoidf_(z));
}

// ---------------------------------------------------------------------------
// 9. Final: silu(bn2(y2) + bns(ysc)), NHWC fp32 -> NCHW fp32 via LDS transpose
// ---------------------------------------------------------------------------
__global__ __launch_bounds__(256)
void k_final(const float* __restrict__ y2, const float* __restrict__ ysc,
             const float* __restrict__ sc2, const float* __restrict__ sh2,
             const float* __restrict__ scs, const float* __restrict__ shs,
             float* __restrict__ out) {
  __shared__ float t[32][33];
  int n = blockIdx.z, c0 = blockIdx.y << 5, hw0 = blockIdx.x << 5;
  int tx = threadIdx.x & 31, ty = threadIdx.x >> 5;
#pragma unroll
  for (int j = 0; j < 4; ++j) {
    int pp = (ty << 2) + j;
    size_t idx = ((size_t)((n << 12) + hw0 + pp) << 8) + c0 + tx;
    int c = c0 + tx;
    float z = y2[idx] * sc2[c] + sh2[c] + ysc[idx] * scs[c] + shs[c];
    t[tx][pp] = z * sigmoidf_(z);
  }
  __syncthreads();
#pragma unroll
  for (int j = 0; j < 4; ++j) {
    int cc = (ty << 2) + j;
    out[((size_t)(n * 256 + c0 + cc) << 12) + hw0 + tx] = t[cc][tx];
  }
}

// ---------------------------------------------------------------------------
extern "C" void kernel_launch(void* const* d_in, const int* in_sizes, int n_in,
                              void* d_out, int out_size, void* d_ws, size_t ws_size,
                              hipStream_t stream) {
  const float* x     = (const float*)d_in[0];
  const float* dce   = (const float*)d_in[1];
  const float* w_dce = (const float*)d_in[2];
  const float* b_dce = (const float*)d_in[3];
  const float* w_ch  = (const float*)d_in[4];
  const float* w_shr = (const float*)d_in[5];
  const float* b_shr = (const float*)d_in[6];
  const float* w_exp = (const float*)d_in[7];
  const float* b_exp = (const float*)d_in[8];
  const float* w_c1  = (const float*)d_in[9];
  const float* g1    = (const float*)d_in[10];
  const float* be1   = (const float*)d_in[11];
  const float* w_c2  = (const float*)d_in[12];
  const float* g2    = (const float*)d_in[13];
  const float* be2   = (const float*)d_in[14];
  const float* w_s   = (const float*)d_in[15];
  const float* gs    = (const float*)d_in[16];
  const float* bes   = (const float*)d_in[17];
  (void)in_sizes; (void)n_in; (void)out_size; (void)ws_size;

  char* ws = (char*)d_ws;
  size_t off = 0;
  auto alloc = [&](size_t bytes) -> void* {
    void* pp = ws + off;
    off += (bytes + 255) & ~(size_t)255;
    return pp;
  };
  float* proj = (float*)alloc(16 * 256 * 4);
  float* spat = (float*)alloc(16 * 256 * 4);
  float* mod  = (float*)alloc(16 * 256 * 4);
  u16*   wb1  = (u16*)alloc((size_t)256 * 2304 * 2);
  u16*   wb2  = (u16*)alloc(65536 * 2);
  u16*   wbs  = (u16*)alloc(65536 * 2);
  u16*   xm   = (u16*)alloc((size_t)65536 * 256 * 2);
  float* y1   = (float*)alloc((size_t)65536 * 256 * 4);
  u16*   y1a  = (u16*)alloc((size_t)65536 * 256 * 2);
  float* y2   = (float*)alloc((size_t)65536 * 256 * 4);
  float* ysc  = (float*)alloc((size_t)65536 * 256 * 4);
  float* ps   = (float*)alloc(256 * 256 * 4);
  float* pq   = (float*)alloc(256 * 256 * 4);
  float* sc1  = (float*)alloc(1024);
  float* sh1  = (float*)alloc(1024);
  float* sc2  = (float*)alloc(1024);
  float* sh2  = (float*)alloc(1024);
  float* scs  = (float*)alloc(1024);
  float* shs  = (float*)alloc(1024);

  k_dce<<<16, 256, 0, stream>>>(dce, w_dce, b_dce, proj);
  k_spatial<<<4096, 256, 0, stream>>>(x, w_ch, spat);
  k_gate<<<16, 256, 0, stream>>>(proj, spat, w_shr, b_shr, w_exp, b_exp, mod);
  k_prep<<<2304, 256, 0, stream>>>(w_c1, w_c2, w_s, wb1, wb2, wbs);
  k_xm<<<dim3(128, 8, 16), 256, 0, stream>>>(x, mod, xm);

  k_gemm<<<dim3(512, 2), 256, 0, stream>>>(xm, wb1, y1, 9);     // conv 3x3
  k_statsp<<<256, 256, 0, stream>>>(y1, ps, pq);
  k_statsf<<<256, 256, 0, stream>>>(ps, pq, g1, be1, sc1, sh1);
  k_bnact<<<65536, 256, 0, stream>>>(y1, sc1, sh1, y1a);

  k_gemm<<<dim3(512, 2), 256, 0, stream>>>(y1a, wb2, y2, 1);    // conv 1x1
  k_gemm<<<dim3(512, 2), 256, 0, stream>>>(xm, wbs, ysc, 1);    // shortcut 1x1

  k_statsp<<<256, 256, 0, stream>>>(y2, ps, pq);
  k_statsf<<<256, 256, 0, stream>>>(ps, pq, g2, be2, sc2, sh2);
  k_statsp<<<256, 256, 0, stream>>>(ysc, ps, pq);
  k_statsf<<<256, 256, 0, stream>>>(ps, pq, gs, bes, scs, shs);

  k_final<<<dim3(128, 8, 16), 256, 0, stream>>>(y2, ysc, sc2, sh2, scs, shs,
                                                (float*)d_out);
}
